// FastSpeech_45380624449549
// MI455X (gfx1250) — compile-verified, matrix operand order checked by reading, not run on hardware
//
#include <hip/hip_runtime.h>
#include <hip/hip_bf16.h>
#include <cstdint>
#include <cstddef>

// ---------------------------------------------------------------------------
// FastSpeech forward on MI455X (gfx1250, wave32, WMMA).
// GEMM-like work (QKV/out-proj/conv1d/conv2/mel, Q.K^T, P.V) runs on
// v_wmma_f32_16x16x32_bf16 with fp32 accumulation. Weights are repacked once
// per launch to bf16 [kw][N][K0] and streamed into LDS with
// global_load_async_to_lds_b128 (ASYNCcnt); activations are converted
// fp32->bf16 while staging through LDS.
// ---------------------------------------------------------------------------

typedef __attribute__((ext_vector_type(8)))  float  v8f;
typedef __attribute__((ext_vector_type(16))) __bf16 v16bf;

#define DEV __device__ __forceinline__

static constexpr int Bsz  = 16;
static constexpr int Lsz  = 128;
static constexpr int Tsz  = 1024;
static constexpr int Dsz  = 256;
static constexpr int Hh   = 2;
static constexpr int HD   = 128;   // head dim
static constexpr int NMELSsz = 80;

// fp32 -> bf16 round-to-nearest-even
DEV unsigned short f2bf(float f) {
  unsigned u = __float_as_uint(f);
  u += 0x7FFFu + ((u >> 16) & 1u);
  return (unsigned short)(u >> 16);
}

// Async copy 16 bytes global -> LDS (gfx1250 ASYNCcnt path).
// Generic LDS pointer low 32 bits == LDS byte offset (LDS aperture low = 0).
DEV void async_g2l_b128(void* lds_dst, const void* gsrc) {
  asm volatile("global_load_async_to_lds_b128 %0, %1, off"
               :: "v"((unsigned)(uintptr_t)lds_dst), "v"(gsrc)
               : "memory");
}
DEV void wait_asynccnt0() {
  asm volatile("s_wait_asynccnt 0x0" ::: "memory");
}

// Load one 16x32 bf16 WMMA operand fragment from an LDS tile stored
// row-major with a stride of 40 halfs (80B rows -> 16B aligned, conflict-free).
// Wave32 layout: lanes 0-15 hold row (lane), K 0..7 & 16..23;
//                lanes 16-31 hold row (lane-16), K 8..15 & 24..31.
DEV v16bf frag_lds(const unsigned short* base, int row0, int lane) {
  int r  = row0 + (lane & 15);
  int kk = (lane >> 4) << 3;            // 0 or 8 halfs
  const unsigned short* p = base + r * 40 + kk;
  union { v16bf v; uint4 u[2]; } f;
  f.u[0] = *reinterpret_cast<const uint4*>(p);        // K kk..kk+7
  f.u[1] = *reinterpret_cast<const uint4*>(p + 16);   // K kk+16..kk+23
  return f.v;
}

// Repack fp32 weight [N, K0, kw] -> bf16 [kw][N][K0] (contiguous per tap).
__global__ __launch_bounds__(256) void wpack_k(const float* __restrict__ w,
                                               unsigned short* __restrict__ o,
                                               int N, int K0, int kw) {
  size_t total = (size_t)N * K0 * kw;
  size_t i = (size_t)blockIdx.x * 256 + threadIdx.x;
  if (i >= total) return;
  size_t c  = i % (size_t)K0;
  size_t nn = (i / (size_t)K0) % (size_t)N;
  size_t kc = i / ((size_t)K0 * (size_t)N);
  o[i] = f2bf(w[(nn * (size_t)K0 + c) * (size_t)kw + kc]);
}

// ---------------------------------------------------------------------------
// Generic WMMA GEMM / conv1d kernel.
//   C[b,s,n] = sum_kc sum_c A[b, s+kc-pad, c] * W[kc][n][c]  (+bias, mask, relu)
// A : [NB*S, K0] fp32, Wb : bf16 [kw][N][K0], C : [NB*S, N] fp32.
// kw==1, pad==0  =>  plain C = A @ W^T + bias.
// Workgroup tile 64(M) x 128(N), K-step 32, 8 waves each computing 32x32.
// ---------------------------------------------------------------------------
__global__ __launch_bounds__(256) void gemm_conv_wmma(
    const float* __restrict__ A, const unsigned short* __restrict__ Wb,
    const float* __restrict__ bias, float* __restrict__ C,
    const float* __restrict__ rowmask,
    int NB, int S, int K0, int N, int kw, int pad, int relu) {
  __shared__ __align__(16) unsigned short As[64 * 40];
  __shared__ __align__(16) unsigned short Bs[128 * 40];
  const int tid = threadIdx.x, lane = tid & 31, wave = tid >> 5;
  const int wm = wave & 1, wn = wave >> 1;
  const int m0 = blockIdx.x * 64, n0 = blockIdx.y * 128;
  const int M = NB * S;
  v8f a00{}, a01{}, a10{}, a11{};
  for (int kc = 0; kc < kw; ++kc) {
    for (int k0 = 0; k0 < K0; k0 += 32) {
      __syncthreads();
      // ---- stage W tile (128 x 32) bf16: async copy global -> LDS ----
#pragma unroll
      for (int i = 0; i < 2; ++i) {
        int idx = tid + i * 256;                 // 0..511 (16B chunks)
        int r = idx >> 2, c8 = (idx & 3) * 8;    // row, col in halfs
        int n = n0 + r;
        unsigned short* dst = Bs + r * 40 + c8;
        if (n < N) {
          const unsigned short* wp = Wb + ((size_t)kc * N + n) * K0 + k0 + c8;
          async_g2l_b128(dst, wp);
        } else {
          *reinterpret_cast<uint4*>(dst) = make_uint4(0u, 0u, 0u, 0u);
        }
      }
      // ---- stage A tile (64 x 32) fp32 -> bf16 ----
#pragma unroll
      for (int i = 0; i < 2; ++i) {
        int idx = tid + i * 256;
        int r = idx >> 3, c4 = (idx & 7) * 4;
        int gr = m0 + r;
        int b  = gr / S, s = gr - b * S;
        int sp = s + kc - pad;
        float4 v = make_float4(0.f, 0.f, 0.f, 0.f);
        if (gr < M && sp >= 0 && sp < S) {
          const float* ap = A + (size_t)(b * S + sp) * K0 + k0 + c4;
          v = *reinterpret_cast<const float4*>(ap);
          if (k0 + 32 < K0) __builtin_prefetch(ap + 32, 0, 1);  // global_prefetch
        }
        unsigned short* d = As + r * 40 + c4;
        d[0] = f2bf(v.x); d[1] = f2bf(v.y); d[2] = f2bf(v.z); d[3] = f2bf(v.w);
      }
      wait_asynccnt0();
      __syncthreads();
      v16bf af0 = frag_lds(As, wm * 32,      lane);
      v16bf af1 = frag_lds(As, wm * 32 + 16, lane);
      v16bf bf0 = frag_lds(Bs, wn * 32,      lane);
      v16bf bf1 = frag_lds(Bs, wn * 32 + 16, lane);
      a00 = __builtin_amdgcn_wmma_f32_16x16x32_bf16(false, af0, false, bf0, (short)0, a00, false, false);
      a01 = __builtin_amdgcn_wmma_f32_16x16x32_bf16(false, af0, false, bf1, (short)0, a01, false, false);
      a10 = __builtin_amdgcn_wmma_f32_16x16x32_bf16(false, af1, false, bf0, (short)0, a10, false, false);
      a11 = __builtin_amdgcn_wmma_f32_16x16x32_bf16(false, af1, false, bf1, (short)0, a11, false, false);
    }
  }
  // ---- epilogue: C/D layout: VGPR e -> M = e (+8 for lanes 16-31), N = lane&15
  const int rofs = (lane >> 4) << 3;
  const int cofs = lane & 15;
#pragma unroll
  for (int i = 0; i < 2; ++i)
#pragma unroll
    for (int j = 0; j < 2; ++j) {
      const v8f& acc = (i == 0) ? (j == 0 ? a00 : a01) : (j == 0 ? a10 : a11);
#pragma unroll
      for (int e = 0; e < 8; ++e) {
        int row = m0 + wm * 32 + i * 16 + rofs + e;
        int col = n0 + wn * 32 + j * 16 + cofs;
        if (row < M && col < N) {
          float v = acc[e] + bias[col];
          if (rowmask) v *= rowmask[row];
          if (relu)    v = fmaxf(v, 0.f);
          C[(size_t)row * N + col] = v;
        }
      }
    }
}

// ---------------------------------------------------------------------------
// Attention scores: S[z, q, k] = (Q_z[q,:] . K_z[k,:]) * scale, key-pad -> -1e9.
// qkv layout [B, S, 3*D]; q at h*HD, k at D + h*HD. z = b*H + h.
// ---------------------------------------------------------------------------
__global__ __launch_bounds__(256) void attn_scores_wmma(
    const float* __restrict__ qkv, const float* __restrict__ nonpad,
    float* __restrict__ scores, int S, float scale) {
  __shared__ __align__(16) unsigned short As[64 * 40];
  __shared__ __align__(16) unsigned short Bs[128 * 40];
  const int tid = threadIdx.x, lane = tid & 31, wave = tid >> 5;
  const int wm = wave & 1, wn = wave >> 1;
  const int m0 = blockIdx.x * 64, n0 = blockIdx.y * 128;
  const int z = blockIdx.z, b = z / Hh, h = z - b * Hh;
  const float* Q  = qkv + (size_t)b * S * (3 * Dsz) + h * HD;
  const float* Kp = qkv + (size_t)b * S * (3 * Dsz) + Dsz + h * HD;
  v8f a00{}, a01{}, a10{}, a11{};
  for (int k0 = 0; k0 < HD; k0 += 32) {
    __syncthreads();
#pragma unroll
    for (int i = 0; i < 2; ++i) {
      int idx = tid + i * 256;
      int r = idx >> 3, c4 = (idx & 7) * 4;
      float4 v = *reinterpret_cast<const float4*>(Q + (size_t)(m0 + r) * (3 * Dsz) + k0 + c4);
      unsigned short* d = As + r * 40 + c4;
      d[0] = f2bf(v.x); d[1] = f2bf(v.y); d[2] = f2bf(v.z); d[3] = f2bf(v.w);
    }
#pragma unroll
    for (int i = 0; i < 4; ++i) {
      int idx = tid + i * 256;
      int r = idx >> 3, c4 = (idx & 7) * 4;
      int kk = n0 + r;
      float4 v = make_float4(0.f, 0.f, 0.f, 0.f);
      if (kk < S) v = *reinterpret_cast<const float4*>(Kp + (size_t)kk * (3 * Dsz) + k0 + c4);
      unsigned short* d = Bs + r * 40 + c4;
      d[0] = f2bf(v.x); d[1] = f2bf(v.y); d[2] = f2bf(v.z); d[3] = f2bf(v.w);
    }
    __syncthreads();
    v16bf af0 = frag_lds(As, wm * 32,      lane);
    v16bf af1 = frag_lds(As, wm * 32 + 16, lane);
    v16bf bf0 = frag_lds(Bs, wn * 32,      lane);
    v16bf bf1 = frag_lds(Bs, wn * 32 + 16, lane);
    a00 = __builtin_amdgcn_wmma_f32_16x16x32_bf16(false, af0, false, bf0, (short)0, a00, false, false);
    a01 = __builtin_amdgcn_wmma_f32_16x16x32_bf16(false, af0, false, bf1, (short)0, a01, false, false);
    a10 = __builtin_amdgcn_wmma_f32_16x16x32_bf16(false, af1, false, bf0, (short)0, a10, false, false);
    a11 = __builtin_amdgcn_wmma_f32_16x16x32_bf16(false, af1, false, bf1, (short)0, a11, false, false);
  }
  const int rofs = (lane >> 4) << 3;
  const int cofs = lane & 15;
#pragma unroll
  for (int i = 0; i < 2; ++i)
#pragma unroll
    for (int j = 0; j < 2; ++j) {
      const v8f& acc = (i == 0) ? (j == 0 ? a00 : a01) : (j == 0 ? a10 : a11);
#pragma unroll
      for (int e = 0; e < 8; ++e) {
        int row = m0 + wm * 32 + i * 16 + rofs + e;   // query
        int col = n0 + wn * 32 + j * 16 + cofs;       // key
        if (col < S) {
          float v = acc[e] * scale;
          if (nonpad[b * S + col] == 0.f) v = -1e9f;
          scores[((size_t)z * S + row) * S + col] = v;
        }
      }
    }
}

// Row softmax over S, in place. One 256-thread block per row.
__global__ __launch_bounds__(256) void softmax_rows(float* __restrict__ s, int S) {
  __shared__ float red[256];
  float* p = s + (size_t)blockIdx.x * S;
  const int t = threadIdx.x;
  float m = -3.4e38f;
  for (int c = t; c < S; c += 256) m = fmaxf(m, p[c]);
  red[t] = m; __syncthreads();
  for (int o = 128; o > 0; o >>= 1) { if (t < o) red[t] = fmaxf(red[t], red[t + o]); __syncthreads(); }
  m = red[0]; __syncthreads();
  float sum = 0.f;
  for (int c = t; c < S; c += 256) { float e = __expf(p[c] - m); p[c] = e; sum += e; }
  red[t] = sum; __syncthreads();
  for (int o = 128; o > 0; o >>= 1) { if (t < o) red[t] += red[t + o]; __syncthreads(); }
  float inv = 1.f / red[0];
  for (int c = t; c < S; c += 256) p[c] *= inv;
}

// ---------------------------------------------------------------------------
// O = P @ V per (b,h): probs [z, S, S] fp32, V from qkv at 2*D + h*HD.
// N = HD = 128 (one column tile). Output attn_out[b, s, h*HD + c].
// ---------------------------------------------------------------------------
__global__ __launch_bounds__(256) void attn_av_wmma(
    const float* __restrict__ probs, const float* __restrict__ qkv,
    float* __restrict__ out, int S) {
  __shared__ __align__(16) unsigned short As[64 * 40];
  __shared__ __align__(16) unsigned short Bs[128 * 40];
  const int tid = threadIdx.x, lane = tid & 31, wave = tid >> 5;
  const int wm = wave & 1, wn = wave >> 1;
  const int m0 = blockIdx.x * 64;
  const int z = blockIdx.z, b = z / Hh, h = z - b * Hh;
  const float* P = probs + (size_t)z * S * S;
  const float* V = qkv + (size_t)b * S * (3 * Dsz) + 2 * Dsz + h * HD;
  v8f a00{}, a01{}, a10{}, a11{};
  for (int k0 = 0; k0 < S; k0 += 32) {
    __syncthreads();
#pragma unroll
    for (int i = 0; i < 2; ++i) {
      int idx = tid + i * 256;
      int r = idx >> 3, c4 = (idx & 7) * 4;
      float4 v = *reinterpret_cast<const float4*>(P + (size_t)(m0 + r) * S + k0 + c4);
      unsigned short* d = As + r * 40 + c4;
      d[0] = f2bf(v.x); d[1] = f2bf(v.y); d[2] = f2bf(v.z); d[3] = f2bf(v.w);
    }
    // stage V^T tile: rows = channel (128), cols = key pos (32)
#pragma unroll
    for (int i = 0; i < 16; ++i) {
      int idx = tid + i * 256;
      int r = idx & 127;       // channel (coalesced across threads)
      int cc = idx >> 7;       // key offset 0..31
      Bs[r * 40 + cc] = f2bf(V[(size_t)(k0 + cc) * (3 * Dsz) + r]);
    }
    __syncthreads();
    v16bf af0 = frag_lds(As, wm * 32,      lane);
    v16bf af1 = frag_lds(As, wm * 32 + 16, lane);
    v16bf bf0 = frag_lds(Bs, wn * 32,      lane);
    v16bf bf1 = frag_lds(Bs, wn * 32 + 16, lane);
    a00 = __builtin_amdgcn_wmma_f32_16x16x32_bf16(false, af0, false, bf0, (short)0, a00, false, false);
    a01 = __builtin_amdgcn_wmma_f32_16x16x32_bf16(false, af0, false, bf1, (short)0, a01, false, false);
    a10 = __builtin_amdgcn_wmma_f32_16x16x32_bf16(false, af1, false, bf0, (short)0, a10, false, false);
    a11 = __builtin_amdgcn_wmma_f32_16x16x32_bf16(false, af1, false, bf1, (short)0, a11, false, false);
  }
  const int rofs = (lane >> 4) << 3;
  const int cofs = lane & 15;
#pragma unroll
  for (int i = 0; i < 2; ++i)
#pragma unroll
    for (int j = 0; j < 2; ++j) {
      const v8f& acc = (i == 0) ? (j == 0 ? a00 : a01) : (j == 0 ? a10 : a11);
#pragma unroll
      for (int e = 0; e < 8; ++e) {
        int row = m0 + wm * 32 + i * 16 + rofs + e;   // query pos
        int col = wn * 32 + j * 16 + cofs;            // channel
        out[((size_t)b * S + row) * Dsz + h * HD + col] = acc[e];
      }
    }
}

// LayerNorm over 256 features (+optional residual, row-mask, relu).
__global__ __launch_bounds__(256) void layernorm_k(
    const float* __restrict__ in, const float* __restrict__ res,
    const float* __restrict__ g, const float* __restrict__ bb,
    const float* __restrict__ rowmask, float* __restrict__ out, int relu) {
  __shared__ float red[256];
  const int row = blockIdx.x, t = threadIdx.x;
  size_t o = (size_t)row * 256 + t;
  float v = in[o];
  if (res) v += res[o];
  red[t] = v; __syncthreads();
  for (int s = 128; s > 0; s >>= 1) { if (t < s) red[t] += red[t + s]; __syncthreads(); }
  float mean = red[0] * (1.f / 256.f);
  __syncthreads();
  float d = v - mean;
  red[t] = d * d; __syncthreads();
  for (int s = 128; s > 0; s >>= 1) { if (t < s) red[t] += red[t + s]; __syncthreads(); }
  float var = red[0] * (1.f / 256.f);
  float y = d * rsqrtf(var + 1e-5f) * g[t] + bb[t];
  if (rowmask) y *= rowmask[row];
  if (relu)    y = fmaxf(y, 0.f);
  out[o] = y;
}

// Duration head: dur[row] = relu(x[row,:].w + b). One wave32 per row.
__global__ __launch_bounds__(256) void dur_linear_k(
    const float* __restrict__ x, const float* __restrict__ w,
    const float* __restrict__ bias, float* __restrict__ dur, int M) {
  const int wave = threadIdx.x >> 5, lane = threadIdx.x & 31;
  const int row = blockIdx.x * 8 + wave;
  if (row >= M) return;
  float s = 0.f;
  for (int k = lane; k < 256; k += 32) s += x[(size_t)row * 256 + k] * w[k];
  for (int o = 16; o > 0; o >>= 1) s += __shfl_xor(s, o, 32);
  if (lane == 0) dur[row] = fmaxf(s + bias[0], 0.f);
}

// Inclusive cumsum of durations (L=128, serial per batch).
__global__ void cumsum_k(const int* __restrict__ durt, int* __restrict__ ends) {
  if (threadIdx.x == 0) {
    const int b = blockIdx.x; int s = 0;
    for (int l = 0; l < Lsz; ++l) { s += durt[b * Lsz + l]; ends[b * Lsz + l] = s; }
  }
}

// Embeddings: x = word_emb[seq] + pos_emb[pos]; mask = seq != 0.
__global__ __launch_bounds__(64) void embed_k(
    const int* __restrict__ seq, const int* __restrict__ pos,
    const float* __restrict__ wemb, const float* __restrict__ pemb,
    float* __restrict__ x, float* __restrict__ mask) {
  const int b = blockIdx.y, l = blockIdx.x, t = threadIdx.x;
  const int sv = seq[b * Lsz + l], pv = pos[b * Lsz + l];
  float4 a = ((const float4*)(wemb + (size_t)sv * 256))[t];
  float4 c = ((const float4*)(pemb + (size_t)pv * 256))[t];
  float4 o = make_float4(a.x + c.x, a.y + c.y, a.z + c.z, a.w + c.w);
  ((float4*)(x + ((size_t)b * Lsz + l) * 256))[t] = o;
  if (t == 0) mask[b * Lsz + l] = (sv != 0) ? 1.f : 0.f;
}

// Length regulator + decoder positional embedding + decoder pad mask.
__global__ __launch_bounds__(64) void lr_k(
    const float* __restrict__ enc, const int* __restrict__ ends,
    const int* __restrict__ mpos, const float* __restrict__ pemb,
    float* __restrict__ x, float* __restrict__ mask) {
  const int b = blockIdx.y, t = blockIdx.x, tt = threadIdx.x;
  const int total = ends[b * Lsz + Lsz - 1];
  int lo = 0, hi = Lsz;
  while (lo < hi) { int mid = (lo + hi) >> 1; if (ends[b * Lsz + mid] > t) hi = mid; else lo = mid + 1; }
  const int pv = mpos[b * Tsz + t];
  float4 base = make_float4(0.f, 0.f, 0.f, 0.f);
  if (t < total) base = ((const float4*)(enc + ((size_t)b * Lsz + lo) * 256))[tt];
  float4 p = ((const float4*)(pemb + (size_t)pv * 256))[tt];
  float4 o = make_float4(base.x + p.x, base.y + p.y, base.z + p.z, base.w + p.w);
  ((float4*)(x + ((size_t)b * Tsz + t) * 256))[tt] = o;
  if (tt == 0) mask[b * Tsz + t] = (pv != 0) ? 1.f : 0.f;
}

// Mel length mask: mask[b,t] = t < max(mel_pos[b,:]).
__global__ __launch_bounds__(256) void melmask_k(const int* __restrict__ mpos,
                                                 float* __restrict__ mask) {
  __shared__ int red[256];
  const int b = blockIdx.x, t = threadIdx.x;
  int m = 0;
  for (int c = t; c < Tsz; c += 256) m = max(m, mpos[b * Tsz + c]);
  red[t] = m; __syncthreads();
  for (int o = 128; o > 0; o >>= 1) { if (t < o) red[t] = max(red[t], red[t + o]); __syncthreads(); }
  const int len = red[0];
  for (int c = t; c < Tsz; c += 256) mask[b * Tsz + c] = (c < len) ? 1.f : 0.f;
}

// ---------------------------------------------------------------------------
// Host orchestration
// ---------------------------------------------------------------------------
static void fft_layer(const float* x, float* xout, int S,
                      const unsigned short* in_wb, const float* in_b,
                      const unsigned short* out_wb, const float* out_b,
                      const unsigned short* w1b, const float* b1,
                      const unsigned short* w2b, const float* b2,
                      const float* lg, const float* lb,
                      const float* mask,
                      float* qkv, float* scores, float* attn,
                      float* ybuf, float* c1, float* c2, hipStream_t st) {
  const int M = Bsz * S;
  const dim3 blk(256);
  const float scale = 0.08838834764831845f;  // 1/sqrt(128)
  gemm_conv_wmma<<<dim3(M / 64, 6), blk, 0, st>>>(x, in_wb, in_b, qkv, nullptr,
                                                  Bsz, S, 256, 768, 1, 0, 0);
  attn_scores_wmma<<<dim3(S / 64, (S + 127) / 128, Bsz * Hh), blk, 0, st>>>(
      qkv, mask, scores, S, scale);
  softmax_rows<<<dim3(Bsz * Hh * S), blk, 0, st>>>(scores, S);
  attn_av_wmma<<<dim3(S / 64, 1, Bsz * Hh), blk, 0, st>>>(scores, qkv, attn, S);
  gemm_conv_wmma<<<dim3(M / 64, 2), blk, 0, st>>>(attn, out_wb, out_b, ybuf, mask,
                                                  Bsz, S, 256, 256, 1, 0, 0);
  gemm_conv_wmma<<<dim3(M / 64, 8), blk, 0, st>>>(ybuf, w1b, b1, c1, nullptr,
                                                  Bsz, S, 256, 1024, 9, 4, 1);  // k=9,pad=4,relu
  gemm_conv_wmma<<<dim3(M / 64, 2), blk, 0, st>>>(c1, w2b, b2, c2, nullptr,
                                                  Bsz, S, 1024, 256, 1, 0, 0);
  layernorm_k<<<dim3(M), blk, 0, st>>>(c2, ybuf, lg, lb, mask, xout, 0);
}

extern "C" void kernel_launch(void* const* d_in, const int* in_sizes, int n_in,
                              void* d_out, int out_size, void* d_ws, size_t ws_size,
                              hipStream_t stream) {
  (void)in_sizes; (void)n_in; (void)out_size; (void)ws_size;
  // ---- inputs (setup_inputs dict order; nested dicts flattened in key order)
  const int*   src_seq  = (const int*)d_in[0];
  const int*   src_pos  = (const int*)d_in[1];
  const int*   mel_pos  = (const int*)d_in[2];
  /* d_in[3] = mel_max_length (== Tsz, compile-time constant here) */
  const int*   len_tgt  = (const int*)d_in[4];
  const float* word_emb = (const float*)d_in[5];
  const float* pos_enc  = (const float*)d_in[6];
  const float* pos_dec  = (const float*)d_in[7];
  const float* enc_w[10]; for (int i = 0; i < 10; ++i) enc_w[i] = (const float*)d_in[8 + i];
  const float* dec_w[10]; for (int i = 0; i < 10; ++i) dec_w[i] = (const float*)d_in[18 + i];
  const float* dp_w1 = (const float*)d_in[28];
  const float* dp_b1 = (const float*)d_in[29];
  const float* dp_l1g = (const float*)d_in[30];
  const float* dp_l1b = (const float*)d_in[31];
  const float* dp_w2 = (const float*)d_in[32];
  const float* dp_b2 = (const float*)d_in[33];
  const float* dp_l2g = (const float*)d_in[34];
  const float* dp_l2b = (const float*)d_in[35];
  const float* dp_lw  = (const float*)d_in[36];
  const float* dp_lb  = (const float*)d_in[37];
  const float* mel_w  = (const float*)d_in[38];
  const float* mel_b  = (const float*)d_in[39];

  // ---- workspace layout (fp32 elements)
  float* ws = (float*)d_ws;
  const size_t SZ_XD = (size_t)Bsz * Tsz * Dsz;            // 4,194,304
  const size_t SZ_QK = (size_t)Bsz * Tsz * 3 * Dsz;        // 12,582,912
  const size_t SZ_C1 = (size_t)Bsz * Tsz * 1024;           // 16,777,216
  const size_t SZ_SC = (size_t)Bsz * Hh * Tsz * Tsz;       // 33,554,432
  const size_t SZ_XE = (size_t)Bsz * Lsz * Dsz;            // 524,288
  float* xdec   = ws;                        // decoder activations [B,T,D]
  float* qkv    = xdec + SZ_XD;
  float* attn   = qkv + SZ_QK;
  float* ybuf   = attn + SZ_XD;
  float* conv1  = ybuf + SZ_XD;
  float* conv2  = conv1 + SZ_C1;
  float* scores = conv2 + SZ_XD;
  float* xenc   = scores + SZ_SC;            // encoder activations [B,L,D]
  float* dpa    = xenc + SZ_XE;
  float* dpb    = dpa + SZ_XE;
  float* mask_e = dpb + SZ_XE;               // [B*L]
  float* mask_d = mask_e + (size_t)Bsz * Lsz;   // [B*T]
  float* mask_m = mask_d + (size_t)Bsz * Tsz;   // [B*T]
  int*   ends   = (int*)(mask_m + (size_t)Bsz * Tsz);  // [B*L]
  // bf16 weight arena (16B aligned: all prior region sizes are multiples of 16B)
  unsigned short* wbp = (unsigned short*)(ends + (size_t)Bsz * Lsz);

  const dim3 blk(256);

  // 0) repack all GEMM weights to bf16 [kw][N][K0]
  auto cvt = [&](const float* src, int N, int K0, int kw) -> const unsigned short* {
    unsigned short* dst = wbp;
    size_t tot = (size_t)N * K0 * kw;
    wpack_k<<<dim3((unsigned)((tot + 255) / 256)), blk, 0, stream>>>(src, dst, N, K0, kw);
    wbp += tot;
    return dst;
  };
  const unsigned short *e_inw[4], *e_outw[4], *e_w1[4], *e_w2[4];
  const unsigned short *d_inw[4], *d_outw[4], *d_w1[4], *d_w2[4];
  for (int i = 0; i < 4; ++i) {
    e_inw[i]  = cvt(enc_w[0] + (size_t)i * 768 * 256, 768, 256, 1);
    e_outw[i] = cvt(enc_w[2] + (size_t)i * 256 * 256, 256, 256, 1);
    e_w1[i]   = cvt(enc_w[4] + (size_t)i * 1024 * 256 * 9, 1024, 256, 9);
    e_w2[i]   = cvt(enc_w[6] + (size_t)i * 256 * 1024, 256, 1024, 1);
  }
  for (int i = 0; i < 4; ++i) {
    d_inw[i]  = cvt(dec_w[0] + (size_t)i * 768 * 256, 768, 256, 1);
    d_outw[i] = cvt(dec_w[2] + (size_t)i * 256 * 256, 256, 256, 1);
    d_w1[i]   = cvt(dec_w[4] + (size_t)i * 1024 * 256 * 9, 1024, 256, 9);
    d_w2[i]   = cvt(dec_w[6] + (size_t)i * 256 * 1024, 256, 1024, 1);
  }
  const unsigned short* dpw1b = cvt(dp_w1, 256, 256, 3);
  const unsigned short* dpw2b = cvt(dp_w2, 256, 256, 3);
  const unsigned short* melwb = cvt(mel_w, NMELSsz, 256, 1);

  // 1) encoder embeddings + pad mask
  embed_k<<<dim3(Lsz, Bsz), 64, 0, stream>>>(src_seq, src_pos, word_emb, pos_enc,
                                             xenc, mask_e);
  // 2) encoder FFT stack (S = 128)
  for (int i = 0; i < 4; ++i) {
    fft_layer(xenc, xenc, Lsz,
              e_inw[i],  enc_w[1] + (size_t)i * 768,
              e_outw[i], enc_w[3] + (size_t)i * 256,
              e_w1[i],   enc_w[5] + (size_t)i * 1024,
              e_w2[i],   enc_w[7] + (size_t)i * 256,
              enc_w[8] + (size_t)i * 256, enc_w[9] + (size_t)i * 256,
              mask_e, qkv, scores, attn, ybuf, conv1, conv2, stream);
  }
  // 3) duration predictor (conv3 -> LN+relu -> conv3 -> LN+relu -> linear+relu)
  const int Me = Bsz * Lsz;
  gemm_conv_wmma<<<dim3(Me / 64, 2), blk, 0, stream>>>(xenc, dpw1b, dp_b1, dpa, nullptr,
                                                       Bsz, Lsz, 256, 256, 3, 1, 0);
  layernorm_k<<<dim3(Me), blk, 0, stream>>>(dpa, nullptr, dp_l1g, dp_l1b, nullptr, dpb, 1);
  gemm_conv_wmma<<<dim3(Me / 64, 2), blk, 0, stream>>>(dpb, dpw2b, dp_b2, dpa, nullptr,
                                                       Bsz, Lsz, 256, 256, 3, 1, 0);
  layernorm_k<<<dim3(Me), blk, 0, stream>>>(dpa, nullptr, dp_l2g, dp_l2b, nullptr, dpb, 1);
  float* dur_out = (float*)d_out + (size_t)Bsz * Tsz * NMELSsz;
  dur_linear_k<<<dim3(Me / 8), blk, 0, stream>>>(dpb, dp_lw, dp_lb, dur_out, Me);
  // 4) length regulation + decoder pos-emb + decoder pad mask
  cumsum_k<<<dim3(Bsz), 32, 0, stream>>>(len_tgt, ends);
  lr_k<<<dim3(Tsz, Bsz), 64, 0, stream>>>(xenc, ends, mel_pos, pos_dec, xdec, mask_d);
  // 5) decoder FFT stack (S = 1024)
  for (int i = 0; i < 4; ++i) {
    fft_layer(xdec, xdec, Tsz,
              d_inw[i],  dec_w[1] + (size_t)i * 768,
              d_outw[i], dec_w[3] + (size_t)i * 256,
              d_w1[i],   dec_w[5] + (size_t)i * 1024,
              d_w2[i],   dec_w[7] + (size_t)i * 256,
              dec_w[8] + (size_t)i * 256, dec_w[9] + (size_t)i * 256,
              mask_d, qkv, scores, attn, ybuf, conv1, conv2, stream);
  }
  // 6) mel projection + length mask -> d_out[:B*T*NMELS]
  melmask_k<<<dim3(Bsz), blk, 0, stream>>>(mel_pos, mask_m);
  gemm_conv_wmma<<<dim3((Bsz * Tsz) / 64, 1), blk, 0, stream>>>(
      xdec, melwb, mel_b, (float*)d_out, mask_m, Bsz, Tsz, 256, NMELSsz, 1, 0, 0);
}
